// WanVAEAttentionBlock_24507083391650
// MI455X (gfx1250) — matrix operand order repaired and checked by reading, not compiled, not run on hardware
//
#include <hip/hip_runtime.h>
#include <hip/hip_bf16.h>
#include <stdint.h>

#define B_   4
#define C_   512
#define HW_  4096
#define G_   8
#define CPG_ 64
#define EPS_ 1e-5f

typedef __attribute__((ext_vector_type(16))) __bf16 v16bf;
typedef __attribute__((ext_vector_type(8)))  float  v8f;

union Frag { v16bf v; uint32_t u[8]; };

__device__ inline v8f zero_v8f() {
    v8f z = {0.f,0.f,0.f,0.f,0.f,0.f,0.f,0.f};
    return z;
}

// A-operand fragment (16x32 bf16). `row` = dword pointer to this lane's M-row,
// contraction contiguous. ISA layout: lanes 0-15: V0..V3 = K 0..7, V4..V7 = K 16..23;
// lanes 16-31: +8.
__device__ inline v16bf load_frag_a(const uint32_t* __restrict__ row, int k0, int lane) {
    const int kb = (lane >> 4) * 8;
    Frag f;
#pragma unroll
    for (int j = 0; j < 4; ++j) {
        f.u[j]     = row[(k0 + kb + 2 * j) >> 1];
        f.u[j + 4] = row[(k0 + 16 + kb + 2 * j) >> 1];
    }
    return f.v;
}

// B-operand fragment (32x16 bf16). `coln` = dword pointer to this lane's N-column
// stored contraction-contiguous ((N,K)-major). ISA layout: lanes 0-15: V0..V3 = K 0..7,
// V4..V7 = K 16..23; lanes 16-31: +8.
__device__ inline v16bf load_frag_b(const uint32_t* __restrict__ coln, int k0, int lane) {
    const int kb = (lane >> 4) * 8;
    Frag f;
#pragma unroll
    for (int j = 0; j < 8; ++j)
        f.u[j] = coln[(k0 + kb + 2 * (j & 3) + ((j >> 2) << 4)) >> 1];
    return f.v;
}

// ---------------- GroupNorm ----------------

__global__ void gn_stats_kernel(const float* __restrict__ x, float* __restrict__ stats) {
    __shared__ float rs[256], rq[256];
    const int bg = blockIdx.x;              // b*8+g, 32 total
    const int b = bg >> 3, g = bg & 7;
    const float* p = x + ((size_t)b * C_ + (size_t)g * CPG_) * HW_;
    float s = 0.f, q = 0.f;
    for (int i = threadIdx.x; i < CPG_ * HW_; i += 256) {
        float v = p[i];
        s += v; q += v * v;
    }
    rs[threadIdx.x] = s; rq[threadIdx.x] = q;
    __syncthreads();
    for (int off = 128; off > 0; off >>= 1) {
        if ((int)threadIdx.x < off) {
            rs[threadIdx.x] += rs[threadIdx.x + off];
            rq[threadIdx.x] += rq[threadIdx.x + off];
        }
        __syncthreads();
    }
    if (threadIdx.x == 0) {
        float inv  = 1.f / (float)(CPG_ * HW_);
        float mean = rs[0] * inv;
        float var  = fmaxf(rq[0] * inv - mean * mean, 0.f);
        stats[bg]      = mean;
        stats[32 + bg] = rsqrtf(var + EPS_);
    }
}

// Normalize + LDS transpose: x (B,C,HW) f32 -> h (B,HW,C) bf16 (token-major).
__global__ void gn_apply_kernel(const float* __restrict__ x, const float* __restrict__ gamma,
                                const float* __restrict__ beta, const float* __restrict__ stats,
                                __hip_bfloat16* __restrict__ ht) {
    __shared__ float tile[64][65];
    const int b  = blockIdx.z;
    const int c0 = blockIdx.y * 64;
    const int p0 = blockIdx.x * 64;
    const int lr = threadIdx.x >> 6;   // 0..3
    const int lc = threadIdx.x & 63;
#pragma unroll
    for (int it = 0; it < 16; ++it) {
        int cl = it * 4 + lr;
        int c  = c0 + cl;
        int g  = c >> 6;
        float mean = stats[b * G_ + g];
        float rstd = stats[32 + b * G_ + g];
        float v = x[((size_t)b * C_ + c) * HW_ + p0 + lc];
        tile[cl][lc] = (v - mean) * rstd * gamma[c] + beta[c];
    }
    __syncthreads();
#pragma unroll
    for (int it = 0; it < 16; ++it) {
        int pl = it * 4 + lr;
        ht[((size_t)b * HW_ + p0 + pl) * C_ + c0 + lc] = __float2bfloat16(tile[lc][pl]);
    }
}

__global__ void cvt_bf16_kernel(const float* __restrict__ s, __hip_bfloat16* __restrict__ d, int n) {
    for (int i = blockIdx.x * blockDim.x + threadIdx.x; i < n; i += gridDim.x * blockDim.x)
        d[i] = __float2bfloat16(s[i]);
}

// ---------------- 512x512 projection GEMMs ----------------
// OUT[m,n] = sum_k W[m,k] * Hin[n,k] + bias[m]   (Hin token-major (HW,C))
// MODE 0: store bf16 transposed (HW,C) via LDS transpose  (Q, K)
// MODE 1: store bf16 normal    (C,HW)                     (V)
// MODE 2: store f32 normal + residual                     (final projection)
template <int MODE>
__global__ void proj_kernel(const __hip_bfloat16* __restrict__ Wb,
                            const __hip_bfloat16* __restrict__ Hin,
                            const float* __restrict__ bias,
                            __hip_bfloat16* __restrict__ outb,
                            float* __restrict__ outf,
                            const float* __restrict__ xres) {
    const int lane = threadIdx.x & 31;
    const int wave = threadIdx.x >> 5;
    const int b = blockIdx.z;
    const int m_base = blockIdx.y * 128 + (wave >> 2) * 64;  // 2 waves in M
    const int n_base = blockIdx.x * 128 + (wave & 3) * 32;   // 4 waves in N

    const uint32_t* Wd = (const uint32_t*)Wb;
    const uint32_t* Hd = (const uint32_t*)Hin + (size_t)b * HW_ * (C_ / 2);

    v8f acc[4][2];
#pragma unroll
    for (int mi = 0; mi < 4; ++mi)
#pragma unroll
        for (int ni = 0; ni < 2; ++ni) acc[mi][ni] = zero_v8f();

    const uint32_t* arow[4];
    const uint32_t* bcol[2];
#pragma unroll
    for (int mi = 0; mi < 4; ++mi)
        arow[mi] = Wd + (size_t)(m_base + mi * 16 + (lane & 15)) * (C_ / 2);
#pragma unroll
    for (int ni = 0; ni < 2; ++ni)
        bcol[ni] = Hd + (size_t)(n_base + ni * 16 + (lane & 15)) * (C_ / 2);

    // software-pipelined K loop: fragments for k+32 are in flight while wmma
    // consumes fragments for k (avoids full s_wait_loadcnt 0 drains before wmma)
    v16bf a_cur[4], b_cur[2];
#pragma unroll
    for (int mi = 0; mi < 4; ++mi) a_cur[mi] = load_frag_a(arow[mi], 0, lane);
#pragma unroll
    for (int ni = 0; ni < 2; ++ni) b_cur[ni] = load_frag_b(bcol[ni], 0, lane);

#pragma unroll 4
    for (int k0 = 0; k0 < C_; k0 += 32) {
        const int k1 = k0 + 32;
        v16bf a_nxt[4], b_nxt[2];
        if (k1 < C_) {
#pragma unroll
            for (int mi = 0; mi < 4; ++mi) a_nxt[mi] = load_frag_a(arow[mi], k1, lane);
#pragma unroll
            for (int ni = 0; ni < 2; ++ni) b_nxt[ni] = load_frag_b(bcol[ni], k1, lane);
        }
#pragma unroll
        for (int mi = 0; mi < 4; ++mi)
#pragma unroll
            for (int ni = 0; ni < 2; ++ni)
                acc[mi][ni] = __builtin_amdgcn_wmma_f32_16x16x32_bf16(
                    false, a_cur[mi], false, b_cur[ni], (short)0, acc[mi][ni], false, false);
        if (k1 < C_) {
#pragma unroll
            for (int mi = 0; mi < 4; ++mi) a_cur[mi] = a_nxt[mi];
#pragma unroll
            for (int ni = 0; ni < 2; ++ni) b_cur[ni] = b_nxt[ni];
        }
    }

    const int hi = lane >> 4;
    if constexpr (MODE == 0) {
        // transpose the wave's 64(M) x 32(N) tile through LDS so the global
        // store is 128B-contiguous per token instead of scattered 2B writes
        __shared__ __hip_bfloat16 tsh[8][32][72];   // 72: 16B-aligned rows, no 64-bank stride
#pragma unroll
        for (int mi = 0; mi < 4; ++mi)
#pragma unroll
            for (int ni = 0; ni < 2; ++ni)
#pragma unroll
                for (int j = 0; j < 8; ++j) {
                    int lm = mi * 16 + j + 8 * hi;          // channel in tile (0..63)
                    int ln = ni * 16 + (lane & 15);         // token in tile   (0..31)
                    tsh[wave][ln][lm] =
                        __float2bfloat16(acc[mi][ni][j] + bias[m_base + lm]);
                }
        __syncthreads();
        const uint32_t* src = (const uint32_t*)&tsh[wave][lane][0];
        uint32_t* dst = (uint32_t*)(outb + (size_t)b * HW_ * C_ +
                                    (size_t)(n_base + lane) * C_ + m_base);
#pragma unroll
        for (int i = 0; i < 32; ++i) dst[i] = src[i];
    } else {
#pragma unroll
        for (int mi = 0; mi < 4; ++mi)
#pragma unroll
            for (int ni = 0; ni < 2; ++ni)
#pragma unroll
                for (int j = 0; j < 8; ++j) {
                    int row = m_base + mi * 16 + j + 8 * hi;
                    int col = n_base + ni * 16 + (lane & 15);
                    float v = acc[mi][ni][j] + bias[row];
                    if (MODE == 1) {
                        outb[(size_t)b * C_ * HW_ + (size_t)row * HW_ + col] =
                            __float2bfloat16(v);
                    } else {
                        size_t idx = (size_t)b * C_ * HW_ + (size_t)row * HW_ + col;
                        outf[idx] = xres[idx] + v;
                    }
                }
    }
}

// ---------------- Streaming (flash) attention ----------------
// Block: 16 queries, 8 waves. Per chunk of 128 keys: each wave computes one 16x16
// score subtile (K=512 contraction), online softmax parallel across all 256
// threads, then each wave accumulates its 64-channel slice of O with
// P (bf16, LDS) x V (L2-resident).
__global__ void attn_kernel(const __hip_bfloat16* __restrict__ qt,
                            const __hip_bfloat16* __restrict__ kt,
                            const __hip_bfloat16* __restrict__ vcm,
                            __hip_bfloat16* __restrict__ ot) {
    __shared__ float sS[16][132];               // padded rows: spread banks
    __shared__ __hip_bfloat16 sP[16][136];      // padded, 16B-aligned rows
    __shared__ float sRed[16][17];
    __shared__ float sM[16], sL[16], sA[16];

    const int lane = threadIdx.x & 31;
    const int wave = threadIdx.x >> 5;
    const int hi = lane >> 4;
    const int b  = blockIdx.y;
    const int q0 = blockIdx.x * 16;
    const int r16 = threadIdx.x >> 4;   // softmax: row owned by this thread group
    const int c16 = threadIdx.x & 15;   // softmax: 8-wide column slice

    const uint32_t* Qd = (const uint32_t*)qt  + (size_t)b * HW_ * (C_ / 2);
    const uint32_t* Kd = (const uint32_t*)kt  + (size_t)b * HW_ * (C_ / 2);
    const uint32_t* Vd = (const uint32_t*)vcm + (size_t)b * C_ * (HW_ / 2);

    if (threadIdx.x < 16) { sM[threadIdx.x] = -1e30f; sL[threadIdx.x] = 0.f; }

    // Preload all 16 Q A-fragments (K=512) -- reused across all 32 key chunks.
    const uint32_t* qrow = Qd + (size_t)(q0 + (lane & 15)) * (C_ / 2);
    v16bf qf[16];
#pragma unroll
    for (int ks = 0; ks < 16; ++ks) qf[ks] = load_frag_a(qrow, ks * 32, lane);

    // this wave's 4 V columns (channels wave*64 .. +63), fixed for all chunks
    const uint32_t* vcol[4];
#pragma unroll
    for (int t = 0; t < 4; ++t)
        vcol[t] = Vd + (size_t)(wave * 64 + t * 16 + (lane & 15)) * (HW_ / 2);

    v8f oacc[4];
#pragma unroll
    for (int t = 0; t < 4; ++t) oacc[t] = zero_v8f();

    __syncthreads();

    const float scale = 0.044194173824159216f;  // 512^-0.5

    for (int ch = 0; ch < 32; ++ch) {
        // --- scores: this wave owns keys [ch*128 + wave*16, +16) ---
        const uint32_t* kcol = Kd + (size_t)(ch * 128 + wave * 16 + (lane & 15)) * (C_ / 2);
        v8f s = zero_v8f();
        v16bf kf = load_frag_b(kcol, 0, lane);
#pragma unroll
        for (int ks = 0; ks < 16; ++ks) {
            v16bf kfn;
            if (ks < 15) kfn = load_frag_b(kcol, (ks + 1) * 32, lane);  // prefetch
            s = __builtin_amdgcn_wmma_f32_16x16x32_bf16(false, qf[ks], false, kf,
                                                        (short)0, s, false, false);
            if (ks < 15) kf = kfn;
        }
#pragma unroll
        for (int j = 0; j < 8; ++j)
            sS[j + 8 * hi][wave * 16 + (lane & 15)] = s[j] * scale;
        __syncthreads();

        // --- online softmax, parallel over all 256 threads ---
        {   // stage 1: per-slice max (each thread: 8 of 128 columns of its row)
            float pmax = -1e30f;
#pragma unroll
            for (int i = 0; i < 8; ++i) pmax = fmaxf(pmax, sS[r16][c16 * 8 + i]);
            sRed[r16][c16] = pmax;
        }
        __syncthreads();
        if (threadIdx.x < 16) {   // stage 2: row max -> new running max, alpha
            const int r = threadIdx.x;
            float tmax = sRed[r][0];
#pragma unroll
            for (int i = 1; i < 16; ++i) tmax = fmaxf(tmax, sRed[r][i]);
            float mNew = fmaxf(sM[r], tmax);
            sA[r] = __expf(sM[r] - mNew);
            sM[r] = mNew;
        }
        __syncthreads();
        {   // stage 3: exponentials + bf16 P + partial sums
            float mNew = sM[r16];
            float psum = 0.f;
#pragma unroll
            for (int i = 0; i < 8; ++i) {
                float p = __expf(sS[r16][c16 * 8 + i] - mNew);
                psum += p;
                sP[r16][c16 * 8 + i] = __float2bfloat16(p);
            }
            sRed[r16][c16] = psum;
        }
        __syncthreads();
        if (threadIdx.x < 16) {   // stage 4: running denominator
            const int r = threadIdx.x;
            float sum = 0.f;
#pragma unroll
            for (int i = 0; i < 16; ++i) sum += sRed[r][i];
            sL[r] = sL[r] * sA[r] + sum;
        }
        __syncthreads();

        // --- rescale running O, then O += P x V; this wave owns channels [wave*64,+64) ---
#pragma unroll
        for (int t = 0; t < 4; ++t)
#pragma unroll
            for (int j = 0; j < 8; ++j) oacc[t][j] *= sA[j + 8 * hi];

        const uint32_t* prow = (const uint32_t*)&sP[lane & 15][0];
#pragma unroll
        for (int kk = 0; kk < 128; kk += 32) {
            v16bf vf[4];
#pragma unroll
            for (int t = 0; t < 4; ++t)
                vf[t] = load_frag_b(vcol[t], ch * 128 + kk, lane);   // issue all loads first
            v16bf pf = load_frag_a(prow, kk, lane);
#pragma unroll
            for (int t = 0; t < 4; ++t)
                oacc[t] = __builtin_amdgcn_wmma_f32_16x16x32_bf16(false, pf, false, vf[t],
                                                                  (short)0, oacc[t], false, false);
        }
        __syncthreads();  // protect sS/sP for next chunk
    }

#pragma unroll
    for (int t = 0; t < 4; ++t) {
#pragma unroll
        for (int j = 0; j < 8; ++j) {
            int m = j + 8 * hi;
            float v = oacc[t][j] / sL[m];
            ot[(size_t)b * HW_ * C_ + (size_t)(q0 + m) * C_ + wave * 64 + t * 16 + (lane & 15)] =
                __float2bfloat16(v);
        }
    }
}

// ---------------- launch ----------------

static constexpr size_t SZ_W    = (size_t)C_ * C_ * 2;          // 512 KB per weight (bf16)
static constexpr size_t SZ_BHWC = (size_t)B_ * HW_ * C_ * 2;    // 16 MB (bf16)
static constexpr size_t OFF_STATS = 0;
static constexpr size_t OFF_WQ = 1024;
static constexpr size_t OFF_WK = OFF_WQ + SZ_W;
static constexpr size_t OFF_WV = OFF_WK + SZ_W;
static constexpr size_t OFF_WO = OFF_WV + SZ_W;
static constexpr size_t OFF_H  = OFF_WO + SZ_W;
static constexpr size_t OFF_Q  = OFF_H + SZ_BHWC;
static constexpr size_t OFF_K  = OFF_Q + SZ_BHWC;
static constexpr size_t OFF_V  = OFF_K + SZ_BHWC;
static constexpr size_t OFF_O  = OFF_V + SZ_BHWC;   // total ~86 MB workspace

extern "C" void kernel_launch(void* const* d_in, const int* in_sizes, int n_in,
                              void* d_out, int out_size, void* d_ws, size_t ws_size,
                              hipStream_t stream) {
    const float* x     = (const float*)d_in[0];
    const float* gamma = (const float*)d_in[1];
    const float* beta  = (const float*)d_in[2];
    const float* wq = (const float*)d_in[3];
    const float* bq = (const float*)d_in[4];
    const float* wk = (const float*)d_in[5];
    const float* bk = (const float*)d_in[6];
    const float* wv = (const float*)d_in[7];
    const float* bv = (const float*)d_in[8];
    const float* wo = (const float*)d_in[9];
    const float* bo = (const float*)d_in[10];
    float* out = (float*)d_out;

    char* ws = (char*)d_ws;
    float*          stats = (float*)(ws + OFF_STATS);
    __hip_bfloat16* wqb = (__hip_bfloat16*)(ws + OFF_WQ);
    __hip_bfloat16* wkb = (__hip_bfloat16*)(ws + OFF_WK);
    __hip_bfloat16* wvb = (__hip_bfloat16*)(ws + OFF_WV);
    __hip_bfloat16* wob = (__hip_bfloat16*)(ws + OFF_WO);
    __hip_bfloat16* ht  = (__hip_bfloat16*)(ws + OFF_H);
    __hip_bfloat16* qtb = (__hip_bfloat16*)(ws + OFF_Q);
    __hip_bfloat16* ktb = (__hip_bfloat16*)(ws + OFF_K);
    __hip_bfloat16* vbm = (__hip_bfloat16*)(ws + OFF_V);
    __hip_bfloat16* otb = (__hip_bfloat16*)(ws + OFF_O);

    gn_stats_kernel<<<32, 256, 0, stream>>>(x, stats);
    gn_apply_kernel<<<dim3(HW_ / 64, C_ / 64, B_), 256, 0, stream>>>(x, gamma, beta, stats, ht);

    cvt_bf16_kernel<<<256, 256, 0, stream>>>(wq, wqb, C_ * C_);
    cvt_bf16_kernel<<<256, 256, 0, stream>>>(wk, wkb, C_ * C_);
    cvt_bf16_kernel<<<256, 256, 0, stream>>>(wv, wvb, C_ * C_);
    cvt_bf16_kernel<<<256, 256, 0, stream>>>(wo, wob, C_ * C_);

    dim3 ggrid(HW_ / 128, C_ / 128, B_);   // (32,4,4)
    proj_kernel<0><<<ggrid, 256, 0, stream>>>(wqb, ht, bq, qtb, nullptr, nullptr);
    proj_kernel<0><<<ggrid, 256, 0, stream>>>(wkb, ht, bk, ktb, nullptr, nullptr);
    proj_kernel<1><<<ggrid, 256, 0, stream>>>(wvb, ht, bv, vbm, nullptr, nullptr);

    attn_kernel<<<dim3(HW_ / 16, B_), 256, 0, stream>>>(qtb, ktb, vbm, otb);

    proj_kernel<2><<<ggrid, 256, 0, stream>>>(wob, otb, bo, nullptr, out, x);
}
